// TransformerBlock_55740085567778
// MI455X (gfx1250) — matrix operand order repaired and checked by reading, not compiled, not run on hardware
//
#include <hip/hip_runtime.h>
#include <hip/hip_bf16.h>
#include <math.h>

#define DEV __device__ __forceinline__

typedef __attribute__((ext_vector_type(16))) __bf16 v16bf;
typedef __attribute__((ext_vector_type(8)))  __bf16 v8bf;
typedef __attribute__((ext_vector_type(8)))  float  v8f;

union BF16x16 { v16bf v; v8bf h[2]; unsigned int u[8]; };
union F32x8   { v8f v; float4 q[2]; float f[8]; };

// sizes
static constexpr int MM   = 16;     // rows
static constexpr int NN   = 2048;   // hidden
static constexpr int DD   = 128;    // head dim
static constexpr int HH   = 16;     // heads
static constexpr int PP   = 8192;   // cache fill
static constexpr int LL   = PP + MM;       // 8208 attended positions
static constexpr int LPAD = 8224;          // padded to multiple of 32
static constexpr int CMAX = 8448;          // cache MAXLEN
static constexpr int FF   = 8192;          // 4*N

DEV unsigned short f2bf(float x) {        // native RNE f32->bf16
  union { __bf16 h; unsigned short s; } r;
  r.h = (__bf16)x;
  return r.s;
}
DEV unsigned int pk2bf(float lo, float hi) {
  union { __bf16 h[2]; unsigned int u; } r;
  r.h[0] = (__bf16)lo; r.h[1] = (__bf16)hi;
  return r.u;
}
DEV v8f wmma_bf16(const BF16x16& a, const BF16x16& b, v8f c) {
  // D = A(16x32 bf16) x B(32x16 bf16) + C(16x16 f32)
  return __builtin_amdgcn_wmma_f32_16x16x32_bf16(false, a.v, false, b.v,
                                                 (short)0, c, false, false);
}

// ---------------------------------------------------------------------------
// LayerNorm: one block per row (16 rows), 256 threads
// ---------------------------------------------------------------------------
__global__ void ln_kernel(const float* __restrict__ X, const float* __restrict__ g,
                          const float* __restrict__ b, float* __restrict__ outF,
                          unsigned short* __restrict__ outB) {
  const int row = blockIdx.x;
  const int tid = threadIdx.x;
  const float* x = X + (size_t)row * NN;
  __shared__ float red[256];
  float s = 0.f;
  for (int i = tid; i < NN; i += 256) s += x[i];
  red[tid] = s; __syncthreads();
  for (int o = 128; o > 0; o >>= 1) { if (tid < o) red[tid] += red[tid + o]; __syncthreads(); }
  const float mu = red[0] * (1.0f / NN); __syncthreads();
  float v = 0.f;
  for (int i = tid; i < NN; i += 256) { float d = x[i] - mu; v += d * d; }
  red[tid] = v; __syncthreads();
  for (int o = 128; o > 0; o >>= 1) { if (tid < o) red[tid] += red[tid + o]; __syncthreads(); }
  const float rs = rsqrtf(red[0] * (1.0f / NN) + 1e-5f);
  for (int i = tid; i < NN; i += 256) {
    float y = (x[i] - mu) * rs * g[i] + b[i];
    if (outF) outF[(size_t)row * NN + i] = y;
    if (outB) outB[(size_t)row * NN + i] = f2bf(y);
  }
}

// ---------------------------------------------------------------------------
// GEMM: out[16 x Nout] = A[16 x K](bf16) * W[Nout x K](f32, row-major)^T
// one wave per (16x16 output tile, K-segment); partial tiles to ws
// ---------------------------------------------------------------------------
__global__ void gemm16_splitk(const unsigned short* __restrict__ A, int lda,
                              const float* __restrict__ W, int ldw,
                              float* __restrict__ partial,
                              int K, int ntiles, int nseg) {
  const int lane = threadIdx.x & 31;
  const int wid = (blockIdx.x * blockDim.x + threadIdx.x) >> 5;
  if (wid >= ntiles * nseg) return;
  const int tile = wid / nseg;
  const int seg  = wid - tile * nseg;
  const int klen = K / nseg;               // multiple of 32
  const int k0   = seg * klen;
  const int nl   = lane & 15;
  const int asel = (lane < 16) ? 0 : 16;   // A: contiguous k = 16h + s
  const int bsel = (lane < 16) ? 0 : 16;   // B: same contiguous k = 16h + s as A
  const unsigned short* arow = A + (size_t)nl * lda + k0 + asel;
  const float* wrow = W + (size_t)(tile * 16 + nl) * ldw + k0 + bsel;
  F32x8 c{};
#pragma unroll 2
  for (int kk = 0; kk < klen; kk += 32) {
    __builtin_prefetch(wrow + kk + 512, 0, 0);   // global_prefetch_b8, 2KB ahead
    BF16x16 a, b;
    *(uint4*)&a.u[0] = *(const uint4*)(arow + kk);
    *(uint4*)&a.u[4] = *(const uint4*)(arow + kk + 8);
    F32x8 wlo, whi;
    wlo.q[0] = *(const float4*)(wrow + kk);
    wlo.q[1] = *(const float4*)(wrow + kk + 4);
    whi.q[0] = *(const float4*)(wrow + kk + 8);
    whi.q[1] = *(const float4*)(wrow + kk + 12);
    b.h[0] = __builtin_convertvector(wlo.v, v8bf);   // v_cvt_pk_bf16_f32 path
    b.h[1] = __builtin_convertvector(whi.v, v8bf);
    c.v = wmma_bf16(a, b, c.v);
  }
  float* p = partial + (size_t)wid * 256 + lane;
#pragma unroll
  for (int r = 0; r < 8; r++) p[r * 32] = c.f[r];
}

// ---------------------------------------------------------------------------
// Reduce split-K partials; optional exact-GELU, residual, f32/bf16 outputs.
// grid = ntiles, block = 256 (one thread per C-matrix register slot)
// ---------------------------------------------------------------------------
__global__ void reduce_tiles(const float* __restrict__ partial, int nseg,
                             const float* __restrict__ residual,
                             float* __restrict__ outF,
                             unsigned short* __restrict__ outB,
                             int act, int ldOut) {
  const int tile = blockIdx.x;
  const int tid = threadIdx.x;
  float s = 0.f;
  const float* p = partial + (size_t)tile * nseg * 256 + tid;
  for (int gI = 0; gI < nseg; ++gI) s += p[gI * 256];
  const int lane = tid & 31, r = tid >> 5;
  const int row = (lane < 16) ? r : (r + 8);
  const int col = tile * 16 + (lane & 15);
  if (act == 1) s = 0.5f * s * (1.0f + erff(s * 0.70710678118654752f));
  if (residual) s += residual[(size_t)row * ldOut + col];
  if (outF) outF[(size_t)row * ldOut + col] = s;
  if (outB) outB[(size_t)row * ldOut + col] = f2bf(s);
}

// ---------------------------------------------------------------------------
// Attention scores: one wave per (head, 16-wide t-tile). S = Q·K^T / sqrt(D)
// Tiles 0..511 stream cache_K; tile 512 reads freshly-computed k (t = P+m).
// ---------------------------------------------------------------------------
__global__ void scores_kernel(const unsigned short* __restrict__ qB,   // 16 x 2048 bf16
                              const float* __restrict__ cacheK,        // H x CMAX x D
                              const float* __restrict__ kF,            // 16 x 2048 f32
                              float* __restrict__ scores) {            // (H*16) x LPAD
  const int lane = threadIdx.x & 31;
  const int wid = (blockIdx.x * blockDim.x + threadIdx.x) >> 5;
  if (wid >= HH * 513) return;
  const int h  = wid / 513;
  const int tt = wid - h * 513;
  const int nl   = lane & 15;
  const int asel = (lane < 16) ? 0 : 16;
  const int bsel = (lane < 16) ? 0 : 16;
  const unsigned short* arow = qB + (size_t)nl * NN + h * DD + asel;
  const float* brow = (tt < 512)
      ? cacheK + ((size_t)h * CMAX + (size_t)tt * 16 + nl) * DD + bsel
      : kF + (size_t)nl * NN + h * DD + bsel;   // row nl == t - P
  F32x8 c{};
#pragma unroll
  for (int kk = 0; kk < DD; kk += 32) {
    BF16x16 a, b;
    *(uint4*)&a.u[0] = *(const uint4*)(arow + kk);
    *(uint4*)&a.u[4] = *(const uint4*)(arow + kk + 8);
    F32x8 wlo, whi;
    wlo.q[0] = *(const float4*)(brow + kk);
    wlo.q[1] = *(const float4*)(brow + kk + 4);
    whi.q[0] = *(const float4*)(brow + kk + 8);
    whi.q[1] = *(const float4*)(brow + kk + 12);
    b.h[0] = __builtin_convertvector(wlo.v, v8bf);
    b.h[1] = __builtin_convertvector(whi.v, v8bf);
    c.v = wmma_bf16(a, b, c.v);
  }
  const float scale = 0.08838834764831845f;   // 1/sqrt(128)
  const int col = tt * 16 + nl;
#pragma unroll
  for (int r = 0; r < 8; r++) {
    int row = (lane < 16) ? r : (r + 8);
    scores[(size_t)(h * 16 + row) * LPAD + col] = c.f[r] * scale;
  }
}

// ---------------------------------------------------------------------------
// Row softmax over L=8208, emit bf16 weights, zero pad [L, LPAD)
// ---------------------------------------------------------------------------
__global__ void softmax_kernel(const float* __restrict__ scores,
                               unsigned short* __restrict__ wOut) {
  const int row = blockIdx.x;                 // 0..255 (h*16+m)
  const int tid = threadIdx.x;                // 256
  const float* s = scores + (size_t)row * LPAD;
  unsigned short* w = wOut + (size_t)row * LPAD;
  __shared__ float red[256];
  float mx = -3.4e38f;
  for (int t = tid; t < LL; t += 256) mx = fmaxf(mx, s[t]);
  red[tid] = mx; __syncthreads();
  for (int o = 128; o > 0; o >>= 1) { if (tid < o) red[tid] = fmaxf(red[tid], red[tid + o]); __syncthreads(); }
  mx = red[0]; __syncthreads();
  float sum = 0.f;
  for (int t = tid; t < LL; t += 256) sum += expf(s[t] - mx);
  red[tid] = sum; __syncthreads();
  for (int o = 128; o > 0; o >>= 1) { if (tid < o) red[tid] += red[tid + o]; __syncthreads(); }
  const float inv = 1.0f / red[0];
  for (int t = tid; t < LL; t += 256) w[t] = f2bf(expf(s[t] - mx) * inv);
  for (int t = LL + tid; t < LPAD; t += 256) w[t] = 0;
}

// ---------------------------------------------------------------------------
// attn = weights · V : one wave per (head, 16-wide d-tile, t-segment)
// chunks 0..255 stream cache_V; chunk 256 reads fresh v (zero-weighted rows
// clamped in-bounds). Partials to ws, reduced by reduce_tiles.
// ---------------------------------------------------------------------------
__global__ void pv_kernel(const unsigned short* __restrict__ wB,   // 256 x LPAD bf16
                          const float* __restrict__ cacheV,        // H x CMAX x D
                          const float* __restrict__ vF,            // 16 x 2048 f32
                          float* __restrict__ partial, int nseg) {
  const int lane = threadIdx.x & 31;
  const int wid = (blockIdx.x * blockDim.x + threadIdx.x) >> 5;
  if (wid >= HH * 8 * nseg) return;
  const int seg = wid % nseg;
  const int dt  = (wid / nseg) % 8;
  const int h   = wid / (nseg * 8);
  const int NCH = LPAD / 32;                 // 257 chunks of 32 t-steps
  const int cps = (NCH + nseg - 1) / nseg;
  const int c0 = seg * cps;
  const int c1 = (c0 + cps < NCH) ? (c0 + cps) : NCH;
  const int nl   = lane & 15;
  const int asel = (lane < 16) ? 0 : 16;
  const int bsel = (lane < 16) ? 0 : 16;
  const unsigned short* arow = wB + (size_t)(h * 16 + nl) * LPAD + asel;
  const float* vcol = cacheV + (size_t)h * CMAX * DD + dt * 16 + nl;
  F32x8 c{};
  for (int ch = c0; ch < c1; ++ch) {
    const int t0 = ch * 32;
    BF16x16 a, b;
    *(uint4*)&a.u[0] = *(const uint4*)(arow + t0);
    *(uint4*)&a.u[4] = *(const uint4*)(arow + t0 + 8);
    if (ch < 256) {
      const float* base = vcol + (size_t)(t0 + bsel) * DD;
#pragma unroll
      for (int j = 0; j < 8; j++)
        b.u[j] = pk2bf(base[(2 * j) * DD], base[(2 * j + 1) * DD]);
    } else {
      const float* vb = vF + h * DD + dt * 16 + nl;
#pragma unroll
      for (int j = 0; j < 8; j++) {
        int r0 = bsel + 2 * j;     if (r0 > 15) r0 = 15;   // rows >=16 carry zero weight
        int r1 = bsel + 2 * j + 1; if (r1 > 15) r1 = 15;
        b.u[j] = pk2bf(vb[(size_t)r0 * NN], vb[(size_t)r1 * NN]);
      }
    }
    c.v = wmma_bf16(a, b, c.v);
  }
  float* p = partial + (size_t)wid * 256 + lane;
#pragma unroll
  for (int r = 0; r < 8; r++) p[r * 32] = c.f[r];
}

// ---------------------------------------------------------------------------
extern "C" void kernel_launch(void* const* d_in, const int* in_sizes, int n_in,
                              void* d_out, int out_size, void* d_ws, size_t ws_size,
                              hipStream_t stream) {
  (void)in_sizes; (void)n_in; (void)out_size; (void)ws_size;
  const float* X   = (const float*)d_in[0];
  const float* g1  = (const float*)d_in[1];
  const float* b1  = (const float*)d_in[2];
  const float* Wq  = (const float*)d_in[3];
  const float* Wk  = (const float*)d_in[4];
  const float* Wv  = (const float*)d_in[5];
  const float* Wo  = (const float*)d_in[6];
  const float* g2  = (const float*)d_in[7];
  const float* b2  = (const float*)d_in[8];
  const float* Wup = (const float*)d_in[9];
  const float* Wdn = (const float*)d_in[10];
  const float* cK  = (const float*)d_in[11];
  const float* cV  = (const float*)d_in[12];

  char* ws = (char*)d_ws;
  size_t off = 0;
  auto alloc = [&](size_t bytes) -> char* {
    char* p = ws + off;
    off = (off + bytes + 255) & ~(size_t)255;
    return p;
  };
  unsigned short* XnB  = (unsigned short*)alloc((size_t)MM * NN * 2);
  unsigned short* qB   = (unsigned short*)alloc((size_t)MM * NN * 2);
  float*          kFp  = (float*)alloc((size_t)MM * NN * 4);
  float*          vFp  = (float*)alloc((size_t)MM * NN * 4);
  float*          sc   = (float*)alloc((size_t)HH * 16 * LPAD * 4);
  unsigned short* wBf  = (unsigned short*)alloc((size_t)HH * 16 * LPAD * 2);
  unsigned short* atB  = (unsigned short*)alloc((size_t)MM * NN * 2);
  float*          X2   = (float*)alloc((size_t)MM * NN * 4);
  unsigned short* Xn2B = (unsigned short*)alloc((size_t)MM * NN * 2);
  unsigned short* HupB = (unsigned short*)alloc((size_t)MM * FF * 2);
  float*          part = (float*)alloc((size_t)4 * 1024 * 1024);   // split-K partials

  auto gemm = [&](const unsigned short* A, int lda, const float* W, int ldw,
                  int K, int ntiles, int nseg) {
    int waves = ntiles * nseg;
    int blocks = (waves * 32 + 127) / 128;
    gemm16_splitk<<<blocks, 128, 0, stream>>>(A, lda, W, ldw, part, K, ntiles, nseg);
  };

  // ---- LN1 ----
  ln_kernel<<<MM, 256, 0, stream>>>(X, g1, b1, nullptr, XnB);
  // ---- Q, K, V projections ----
  gemm(XnB, NN, Wq, NN, NN, 128, 8);
  reduce_tiles<<<128, 256, 0, stream>>>(part, 8, nullptr, nullptr, qB, 0, NN);
  gemm(XnB, NN, Wk, NN, NN, 128, 8);
  reduce_tiles<<<128, 256, 0, stream>>>(part, 8, nullptr, kFp, nullptr, 0, NN);
  gemm(XnB, NN, Wv, NN, NN, 128, 8);
  reduce_tiles<<<128, 256, 0, stream>>>(part, 8, nullptr, vFp, nullptr, 0, NN);
  // ---- attention ----
  {
    int waves = HH * 513, blocks = (waves * 32 + 127) / 128;
    scores_kernel<<<blocks, 128, 0, stream>>>(qB, cK, kFp, sc);
  }
  softmax_kernel<<<HH * 16, 256, 0, stream>>>(sc, wBf);
  {
    int nseg = 16, waves = HH * 8 * nseg, blocks = (waves * 32 + 127) / 128;
    pv_kernel<<<blocks, 128, 0, stream>>>(wBf, cV, vFp, part, nseg);
    reduce_tiles<<<128, 256, 0, stream>>>(part, nseg, nullptr, nullptr, atB, 0, NN);
  }
  // ---- output projection + residual ----
  gemm(atB, NN, Wo, NN, NN, 128, 8);
  reduce_tiles<<<128, 256, 0, stream>>>(part, 8, X, X2, nullptr, 0, NN);
  // ---- LN2 ----
  ln_kernel<<<MM, 256, 0, stream>>>(X2, g2, b2, nullptr, Xn2B);
  // ---- FFN up + exact GELU ----
  gemm(Xn2B, NN, Wup, NN, NN, 512, 8);
  reduce_tiles<<<512, 256, 0, stream>>>(part, 8, nullptr, nullptr, HupB, 1, FF);
  // ---- FFN down + residual -> out ----
  gemm(HupB, FF, Wdn, FF, FF, 128, 32);
  reduce_tiles<<<128, 256, 0, stream>>>(part, 32, X2, (float*)d_out, nullptr, 0, NN);
}
